// TransformerSRUEncoderLayer_24988119728164
// MI455X (gfx1250) — compile-verified, matrix operand order checked by reading, not run on hardware
//
#include <hip/hip_runtime.h>

#define L_ 2048
#define B_ 4
#define D_ 1024
#define H_ 16
#define HD_ 64
#define FF_ 2048
#define M_ (L_ * B_)              /* 8192 rows when [L,B,*] is flattened */
#define THREE_D (3 * D_)          /* 3072 */
#define STRIDE_L (B_ * THREE_D)   /* 12288 floats between consecutive l in qkv */

typedef _Float16 f16;
typedef __attribute__((ext_vector_type(16))) _Float16 v16h;
typedef __attribute__((ext_vector_type(8)))  float    v8f;

union FragH { v16h v; _Float16 h[16]; unsigned int u[8]; };
union FragF { v8f  v; float f[8]; };

// ---------------------------------------------------------------------------
// CDNA5 async global->LDS copy (ASYNCcnt) with safe fallback
// ---------------------------------------------------------------------------
#if __has_builtin(__builtin_amdgcn_global_load_async_to_lds_b128)
#define HAS_ASYNC_LDS 1
#else
#define HAS_ASYNC_LDS 0
#endif

typedef int v4i_b128 __attribute__((vector_size(16)));

__device__ __forceinline__ void copy16_g2l(const f16* g, f16* l) {
#if HAS_ASYNC_LDS
  typedef __attribute__((address_space(1))) v4i_b128 gv4i;  // global (AS1)
  typedef __attribute__((address_space(3))) v4i_b128 lv4i;  // LDS (AS3)
  __builtin_amdgcn_global_load_async_to_lds_b128(
      (gv4i*)(unsigned long long)(uintptr_t)g,
      (lv4i*)(unsigned int)(uintptr_t)l,
      0, 0);
#else
  *(uint4*)l = *(const uint4*)g;
#endif
}

__device__ __forceinline__ void wait_async0() {
#if HAS_ASYNC_LDS
#if __has_builtin(__builtin_amdgcn_s_wait_asynccnt)
  __builtin_amdgcn_s_wait_asynccnt(0);
#else
  asm volatile("s_wait_asynccnt 0" ::: "memory");
#endif
#endif
}

// ---------------------------------------------------------------------------
// Generic f16 WMMA GEMM: C[M,N](f32) = A[M,K](f16) @ B[K,N](f16) (+bias)(+res)
// Block tile 128x128, wave tile 32x64 (8 WMMA / K-step), double-buffered LDS,
// async A staging, single barrier per K-step.
// ---------------------------------------------------------------------------
#define BM 128
#define BN 128
#define BK 32
#define LDSS 40  // halfs per LDS row (pad to dodge bank conflicts)

// Load a 16x32 f16 fragment from LDS laid out [row][k] (pairs along k).
// ISA layout: lanes 0-15 hold K {0..7,16..23}; lanes 16-31 hold K {8..15,24..31}.
__device__ __forceinline__ FragH frag_lds(const f16* lds, int rowBase) {
  const int lane = threadIdx.x & 31;
  const int hf = lane >> 4;
  const unsigned int* rp =
      (const unsigned int*)(lds + (rowBase + (lane & 15)) * LDSS);
  FragH f;
  const int p0 = hf * 4;
#pragma unroll
  for (int i = 0; i < 4; ++i) {
    f.u[i]     = rp[p0 + i];
    f.u[4 + i] = rp[p0 + 8 + i];
  }
  return f;
}

__global__ __launch_bounds__(256) void gemm_f16_kernel(
    const f16* __restrict__ A, const f16* __restrict__ Bm,
    const float* __restrict__ bias, const float* __restrict__ res,
    float* __restrict__ C, int M, int N, int K) {
  __shared__ __align__(16) f16 ldsA[2][BM * LDSS];  // [row][k]
  __shared__ __align__(16) f16 ldsB[2][BN * LDSS];  // [n][k] (transposed)
  const int tid = threadIdx.x;
  const int lane = tid & 31, wave = tid >> 5;
  const int hf = lane >> 4, ln = lane & 15;
  const int bm = blockIdx.x * BM, bn = blockIdx.y * BN;
  const int wm = (wave >> 1) * 32;  // 0,32,64,96
  const int wn = (wave & 1) * 64;   // 0,64

  uint4 breg[2];

  // A tile: 128 rows x 32 halfs; 2 x 16B chunks per thread, async into LDS.
  auto stageA = [&](int buf, int k0) {
#pragma unroll
    for (int it = 0; it < 2; ++it) {
      int chunk = tid + it * 256;
      int r = chunk >> 2, ck = (chunk & 3) * 8;
      copy16_g2l(A + (size_t)(bm + r) * K + k0 + ck,
                 &ldsA[buf][r * LDSS + ck]);
    }
  };
  // B tile: 32 k-rows x 128 n; load 16B chunks into regs (issue early) ...
  auto loadB = [&](int k0) {
#pragma unroll
    for (int it = 0; it < 2; ++it) {
      int chunk = tid + it * 256;
      int k = chunk >> 4, n0 = (chunk & 15) * 8;
      breg[it] = *(const uint4*)(Bm + (size_t)(k0 + k) * N + bn + n0);
    }
  };
  // ... then scatter transposed into LDS [n][k].
  auto storeB = [&](int buf) {
#pragma unroll
    for (int it = 0; it < 2; ++it) {
      int chunk = tid + it * 256;
      int k = chunk >> 4, n0 = (chunk & 15) * 8;
      union { uint4 q; f16 h[8]; } t;
      t.q = breg[it];
#pragma unroll
      for (int i = 0; i < 8; ++i) ldsB[buf][(n0 + i) * LDSS + k] = t.h[i];
    }
  };

  FragF c[2][4];
#pragma unroll
  for (int i = 0; i < 2; ++i)
#pragma unroll
    for (int j = 0; j < 4; ++j)
#pragma unroll
      for (int r = 0; r < 8; ++r) c[i][j].f[r] = 0.f;

  // Prologue: stage tile 0.
  loadB(0);
  stageA(0, 0);
  storeB(0);
  wait_async0();
  __syncthreads();

  const int nk = K / BK;
  for (int kt = 0; kt < nk; ++kt) {
    const int cur = kt & 1, nxt = cur ^ 1;
    const bool more = (kt + 1 < nk);
    if (more) {               // prefetch next tile into the other buffer
      stageA(nxt, (kt + 1) * BK);
      loadB((kt + 1) * BK);
    }
    // Compute current tile: 2 A frags x 4 B frags -> 8 WMMAs.
    FragH a0 = frag_lds(&ldsA[cur][0], wm);
    FragH a1 = frag_lds(&ldsA[cur][0], wm + 16);
    FragH b[4];
#pragma unroll
    for (int j = 0; j < 4; ++j) b[j] = frag_lds(&ldsB[cur][0], wn + j * 16);
#pragma unroll
    for (int j = 0; j < 4; ++j) {
      c[0][j].v = __builtin_amdgcn_wmma_f32_16x16x32_f16(false, a0.v, false, b[j].v, (short)0, c[0][j].v, false, false);
      c[1][j].v = __builtin_amdgcn_wmma_f32_16x16x32_f16(false, a1.v, false, b[j].v, (short)0, c[1][j].v, false, false);
    }
    if (more) storeB(nxt);
    wait_async0();
    __syncthreads();
  }

#pragma unroll
  for (int i = 0; i < 2; ++i)
#pragma unroll
    for (int j = 0; j < 4; ++j)
#pragma unroll
      for (int r = 0; r < 8; ++r) {
        int row = bm + wm + i * 16 + r + 8 * hf;  // C: VGPR r -> M=r (+8 hi half)
        int col = bn + wn + j * 16 + ln;          // N = lane&15
        float v = c[i][j].f[r];
        if (bias) v += bias[col];
        size_t idx = (size_t)row * N + col;
        if (res) v += res[idx];
        C[idx] = v;
      }
}

// ---------------------------------------------------------------------------
// Flash attention: block = (b,h) x 32 query rows; online softmax over 2048 keys
// ---------------------------------------------------------------------------
#define AQ 32
#define ACH 256

__global__ __launch_bounds__(256) void attn_flash_kernel(
    const float* __restrict__ qkv, f16* __restrict__ o_h) {
  __shared__ float S[AQ][ACH + 4];
  __shared__ float red[256];
  __shared__ float rmax[AQ], rsum[AQ], fac[AQ];

  const int tid = threadIdx.x;
  const int lane = tid & 31, wave = tid >> 5;
  const int hf = lane >> 4, ln = lane & 15;
  const int bh = blockIdx.x;
  const int b = bh >> 4, h = bh & 15;
  const int qb = blockIdx.y * AQ;

  const float* qbase = qkv + (size_t)b * THREE_D + h * HD_;
  const float* kbase = qbase + D_;
  const float* vbase = qbase + 2 * D_;

  // Q A-fragments straight from global (d contiguous), scaled by 1/sqrt(HD)
  FragH qa[2][2];
#pragma unroll
  for (int rb = 0; rb < 2; ++rb) {
    const float* qp = qbase + (size_t)(qb + rb * 16 + ln) * STRIDE_L;
#pragma unroll
    for (int kc = 0; kc < 2; ++kc) {
      int d0 = kc * 32 + hf * 8;
#pragma unroll
      for (int i = 0; i < 8; ++i) {
        qa[rb][kc].h[i]     = (f16)(qp[d0 + i] * 0.125f);
        qa[rb][kc].h[8 + i] = (f16)(qp[d0 + 16 + i] * 0.125f);
      }
    }
  }

  const int rbO = (wave >> 2) * 16;
  const int dcO = (wave & 3) * 16;
  FragF oacc;
#pragma unroll
  for (int r = 0; r < 8; ++r) oacc.f[r] = 0.f;

  if (tid < AQ) { rmax[tid] = -1e30f; rsum[tid] = 0.f; }
  __syncthreads();

  for (int ck = 0; ck < L_ / ACH; ++ck) {
    const int kb0 = ck * ACH;
    // ---- phase 1: S = Q @ K^T for this 256-key chunk (waves split keys) ----
#pragma unroll
    for (int ii = 0; ii < 2; ++ii) {
      int kb16 = wave * 16 + ii * 128;
      const float* kp = kbase + (size_t)(kb0 + kb16 + ln) * STRIDE_L;
      FragH kf[2];
#pragma unroll
      for (int kc = 0; kc < 2; ++kc) {
        int d0 = kc * 32 + hf * 8;
#pragma unroll
        for (int i = 0; i < 8; ++i) {
          kf[kc].h[i]     = (f16)kp[d0 + i];
          kf[kc].h[8 + i] = (f16)kp[d0 + 16 + i];
        }
      }
#pragma unroll
      for (int rb = 0; rb < 2; ++rb) {
        FragF s;
#pragma unroll
        for (int r = 0; r < 8; ++r) s.f[r] = 0.f;
        s.v = __builtin_amdgcn_wmma_f32_16x16x32_f16(false, qa[rb][0].v, false, kf[0].v, (short)0, s.v, false, false);
        s.v = __builtin_amdgcn_wmma_f32_16x16x32_f16(false, qa[rb][1].v, false, kf[1].v, (short)0, s.v, false, false);
#pragma unroll
        for (int r = 0; r < 8; ++r) S[rb * 16 + r + 8 * hf][kb16 + ln] = s.f[r];
      }
    }
    __syncthreads();

    // ---- online softmax update (8 threads per row) ----
    {
      int row = tid >> 3, seg = tid & 7;
      float m = -1e30f;
      for (int i = 0; i < 32; ++i) m = fmaxf(m, S[row][seg * 32 + i]);
      red[tid] = m;
      __syncthreads();
      if (seg == 0) {
        for (int k = 1; k < 8; ++k) m = fmaxf(m, red[tid + k]);
        float om = rmax[row];
        float nm = fmaxf(om, m);
        fac[row] = __expf(om - nm);
        rmax[row] = nm;
      }
      __syncthreads();
      float nm = rmax[row];
      float s = 0.f;
      for (int i = 0; i < 32; ++i) {
        float e = __expf(S[row][seg * 32 + i] - nm);
        S[row][seg * 32 + i] = e;
        s += e;
      }
      red[tid] = s;
      __syncthreads();
      if (seg == 0) {
        float ss = 0.f;
        for (int k = 0; k < 8; ++k) ss += red[tid + k];
        rsum[row] = rsum[row] * fac[row] + ss;
      }
      __syncthreads();
    }

    // ---- phase 2: O = O*fac + P @ V (each wave owns one 16x16 O fragment) ----
#pragma unroll
    for (int r = 0; r < 8; ++r) oacc.f[r] *= fac[rbO + r + 8 * hf];
    for (int kk = 0; kk < ACH; kk += 32) {
      FragH pa;
      {
        int prow = rbO + ln;
        int kc0 = hf * 8;
#pragma unroll
        for (int i = 0; i < 8; ++i) {
          pa.h[i]     = (f16)S[prow][kk + kc0 + i];
          pa.h[8 + i] = (f16)S[prow][kk + 16 + kc0 + i];
        }
      }
      FragH vb;
      {
        const float* vp = vbase + dcO + ln;
        int keyb = kb0 + kk;
#pragma unroll
        for (int i = 0; i < 8; ++i) {
          vb.h[i]     = (f16)vp[(size_t)(keyb + hf * 8 + i) * STRIDE_L];
          vb.h[8 + i] = (f16)vp[(size_t)(keyb + 16 + hf * 8 + i) * STRIDE_L];
        }
      }
      oacc.v = __builtin_amdgcn_wmma_f32_16x16x32_f16(false, pa.v, false, vb.v, (short)0, oacc.v, false, false);
    }
    __syncthreads();
  }

#pragma unroll
  for (int r = 0; r < 8; ++r) {
    int row = rbO + r + 8 * hf;
    float v = oacc.f[r] / rsum[row];
    int l = qb + row;
    o_h[(size_t)(l * B_ + b) * D_ + h * HD_ + dcO + ln] = (f16)v;
  }
}

// ---------------------------------------------------------------------------
// LayerNorm over last dim (D=1024), one block per (l,b) row; optional f16 copy
// ---------------------------------------------------------------------------
__global__ __launch_bounds__(256) void layernorm_kernel(
    const float* __restrict__ in, const float* __restrict__ gamma,
    const float* __restrict__ beta, float* __restrict__ out,
    f16* __restrict__ out_h) {
  __shared__ float red[256], red2[256];
  __shared__ float mu_s, rstd_s;
  const int row = blockIdx.x, tid = threadIdx.x;
  const float* x = in + (size_t)row * D_;
  float s = 0.f, s2 = 0.f;
  for (int i = tid; i < D_; i += 256) { float v = x[i]; s += v; s2 += v * v; }
  red[tid] = s; red2[tid] = s2;
  __syncthreads();
  for (int off = 128; off > 0; off >>= 1) {
    if (tid < off) { red[tid] += red[tid + off]; red2[tid] += red2[tid + off]; }
    __syncthreads();
  }
  if (tid == 0) {
    float mu = red[0] * (1.f / D_);
    float var = red2[0] * (1.f / D_) - mu * mu;
    mu_s = mu;
    rstd_s = rsqrtf(var + 1e-5f);
  }
  __syncthreads();
  float mu = mu_s, rs = rstd_s;
  for (int i = tid; i < D_; i += 256) {
    float v = (x[i] - mu) * rs * gamma[i] + beta[i];
    out[(size_t)row * D_ + i] = v;
    if (out_h) out_h[(size_t)row * D_ + i] = (f16)v;
  }
}

// ---------------------------------------------------------------------------
// SRU recurrence: one thread per (b,ff) channel, sequential over L
// ---------------------------------------------------------------------------
__global__ __launch_bounds__(256) void sru_scan_kernel(
    const float* __restrict__ U, const float* __restrict__ sru_b,
    f16* __restrict__ h_out) {
  const int idx = blockIdx.x * 256 + threadIdx.x;  // 0..B*FF-1
  const int b = idx >> 11;                         // FF_ == 2048
  const int ff = idx & (FF_ - 1);
  const float bf = sru_b[ff], br = sru_b[FF_ + ff];
  float c = 0.f;
  for (int l = 0; l < L_; ++l) {
    const float* u = U + (size_t)(l * B_ + b) * (4 * FF_);
    float u0 = u[ff], u1 = u[FF_ + ff], u2 = u[2 * FF_ + ff], u3 = u[3 * FF_ + ff];
    float f = 1.f / (1.f + __expf(-(u1 + bf)));
    c = f * c + (1.f - f) * u0;
    float r = 1.f / (1.f + __expf(-(u2 + br)));
    float hv = r * c + (1.f - r) * u3;
    h_out[(size_t)(l * B_ + b) * FF_ + ff] = (f16)hv;
  }
}

// ---------------------------------------------------------------------------
// f32 -> f16 weight conversion into [K][N] layout (optional transpose)
// ---------------------------------------------------------------------------
__global__ void cvt_weight_kernel(const float* __restrict__ W,
                                  f16* __restrict__ out, int K, int N,
                                  int transpose) {
  int idx = blockIdx.x * 256 + threadIdx.x;
  if (idx >= K * N) return;
  int k = idx / N, n = idx - k * N;
  float v = transpose ? W[(size_t)n * K + k] : W[(size_t)k * N + n];
  out[idx] = (f16)v;
}

__global__ void cvt_f32_f16_kernel(const float* __restrict__ in,
                                   f16* __restrict__ out, int n) {
  int i = blockIdx.x * 256 + threadIdx.x;
  if (i < n) out[i] = (f16)in[i];
}

// ---------------------------------------------------------------------------
extern "C" void kernel_launch(void* const* d_in, const int* in_sizes, int n_in,
                              void* d_out, int out_size, void* d_ws,
                              size_t ws_size, hipStream_t stream) {
  (void)in_sizes; (void)n_in; (void)out_size; (void)ws_size;
  const float* src        = (const float*)d_in[0];
  const float* attn_in_w  = (const float*)d_in[1];
  const float* attn_in_b  = (const float*)d_in[2];
  const float* attn_out_w = (const float*)d_in[3];
  const float* attn_out_b = (const float*)d_in[4];
  const float* sru_w      = (const float*)d_in[5];
  const float* sru_b      = (const float*)d_in[6];
  const float* lin2_w     = (const float*)d_in[7];
  const float* lin2_b     = (const float*)d_in[8];
  const float* ln1_s      = (const float*)d_in[9];
  const float* ln1_b      = (const float*)d_in[10];
  const float* ln2_s      = (const float*)d_in[11];
  const float* ln2_b      = (const float*)d_in[12];

  char* ws = (char*)d_ws;
  size_t off = 0;
  auto alloc = [&](size_t bytes) -> void* {
    void* p = ws + off;
    off += (bytes + 255) & ~(size_t)255;
    return p;
  };

  f16*   src_h    = (f16*)  alloc((size_t)M_ * D_ * 2);
  f16*   w_qkv_h  = (f16*)  alloc((size_t)D_ * THREE_D * 2);
  f16*   w_out_h  = (f16*)  alloc((size_t)D_ * D_ * 2);
  f16*   w_sru_h  = (f16*)  alloc((size_t)D_ * 4 * FF_ * 2);
  f16*   w_lin2_h = (f16*)  alloc((size_t)FF_ * D_ * 2);
  float* qkv      = (float*)alloc((size_t)M_ * THREE_D * 4);
  f16*   o_h      = (f16*)  alloc((size_t)M_ * D_ * 2);
  float* y1       = (float*)alloc((size_t)M_ * D_ * 4);
  float* x1       = (float*)alloc((size_t)M_ * D_ * 4);
  f16*   x1_h     = (f16*)  alloc((size_t)M_ * D_ * 2);
  float* U        = (float*)alloc((size_t)M_ * 4 * FF_ * 4);
  f16*   h_h      = (f16*)  alloc((size_t)M_ * FF_ * 2);
  float* y2       = (float*)alloc((size_t)M_ * D_ * 4);

  // Weight / input casts (B matrices stored [K][N] f16)
  cvt_weight_kernel<<<(D_ * THREE_D + 255) / 256, 256, 0, stream>>>(attn_in_w, w_qkv_h, D_, THREE_D, 1);
  cvt_weight_kernel<<<(D_ * D_ + 255) / 256, 256, 0, stream>>>(attn_out_w, w_out_h, D_, D_, 1);
  cvt_weight_kernel<<<(D_ * 4 * FF_ + 255) / 256, 256, 0, stream>>>(sru_w, w_sru_h, D_, 4 * FF_, 0);
  cvt_weight_kernel<<<(FF_ * D_ + 255) / 256, 256, 0, stream>>>(lin2_w, w_lin2_h, FF_, D_, 1);
  cvt_f32_f16_kernel<<<(M_ * D_ + 255) / 256, 256, 0, stream>>>(src, src_h, M_ * D_);

  // qkv = src @ in_w^T + in_b
  gemm_f16_kernel<<<dim3(M_ / BM, THREE_D / BN), 256, 0, stream>>>(
      src_h, w_qkv_h, attn_in_b, nullptr, qkv, M_, THREE_D, D_);
  // attention -> o (f16)
  attn_flash_kernel<<<dim3(B_ * H_, L_ / AQ), 256, 0, stream>>>(qkv, o_h);
  // y1 = o @ out_w^T + out_b + src
  gemm_f16_kernel<<<dim3(M_ / BM, D_ / BN), 256, 0, stream>>>(
      o_h, w_out_h, attn_out_b, src, y1, M_, D_, D_);
  // x1 = LN1(y1), also f16 copy for SRU GEMM
  layernorm_kernel<<<M_, 256, 0, stream>>>(y1, ln1_s, ln1_b, x1, x1_h);
  // U = x1 @ sru_w
  gemm_f16_kernel<<<dim3(M_ / BM, (4 * FF_) / BN), 256, 0, stream>>>(
      x1_h, w_sru_h, nullptr, nullptr, U, M_, 4 * FF_, D_);
  // SRU recurrence -> h (f16)
  sru_scan_kernel<<<(B_ * FF_) / 256, 256, 0, stream>>>(U, sru_b, h_h);
  // y2 = h @ lin2_w^T + lin2_b + x1
  gemm_f16_kernel<<<dim3(M_ / BM, D_ / BN), 256, 0, stream>>>(
      h_h, w_lin2_h, lin2_b, x1, y2, M_, D_, FF_);
  // out = LN2(y2)
  layernorm_kernel<<<M_, 256, 0, stream>>>(y2, ln2_s, ln2_b, (float*)d_out, nullptr);
}
